// HeteroDoshaNet_36730560315562
// MI455X (gfx1250) — matrix operand order repaired and checked by reading, not compiled, not run on hardware
//
#include <hip/hip_runtime.h>
#include <math.h>

// ---------------- problem constants (match reference) ----------------
#define NP    100000
#define NS    2000
#define ND    3
#define CP    128
#define CS    64
#define CD    16
#define HID   128
#define HEADS 4
#define HDIM  32          // HID / HEADS
#define NCLS  3
#define EHAS  300000
#define EBEL  100000
#define ESIM  400000
#define BN_EPS 1e-5f
#define GROWS 64          // M-rows per GEMM block (4 wmma tiles per wave)

#define CDIV(a,b) (((a)+(b)-1)/(b))

#if __has_builtin(__builtin_amdgcn_global_load_async_to_lds_b128)
#define HAVE_ASYNC_LDS 1
typedef int vint4_ __attribute__((vector_size(16)));            // matches builtin param
typedef __attribute__((address_space(3))) vint4_ as3_vint4;     // LDS-qualified variant
#endif

typedef __attribute__((ext_vector_type(16))) __bf16 v16bf;
typedef __attribute__((ext_vector_type(8)))  float  v8f;

// ---------------- helpers ----------------
__device__ __forceinline__ __bf16 f2bf(float f) {
  unsigned u = __float_as_uint(f);
  u += 0x7FFFu + ((u >> 16) & 1u);           // round-to-nearest-even
  unsigned short h = (unsigned short)(u >> 16);
  __bf16 r; __builtin_memcpy(&r, &h, 2);
  return r;
}

// monotonic-bit-pattern float atomic max (works for mixed signs, -inf init)
__device__ __forceinline__ void atomicMaxF(float* addr, float v) {
  if (!__builtin_signbitf(v))
    atomicMax((int*)addr, __float_as_int(v));
  else
    atomicMin((unsigned int*)addr, __float_as_uint(v));
}

__device__ __forceinline__ float lrelu02(float x) { return x > 0.f ? x : 0.2f * x; }

// ---------------- conversion kernels ----------------
__global__ void k_f32_to_bf16(const float* __restrict__ s, __bf16* __restrict__ d, int n) {
  int i = blockIdx.x * blockDim.x + threadIdx.x;
  if (i < n) d[i] = f2bf(s[i]);
}

// B[k*N + n] = W[n*K + k]  (W is [N,K] row-major; B is K x N for C = A @ Wt)
__global__ void k_wT_to_bf16(const float* __restrict__ W, __bf16* __restrict__ B, int N, int K) {
  int i = blockIdx.x * blockDim.x + threadIdx.x;
  if (i >= N * K) return;
  int n = i / K, k = i % K;
  B[k * N + n] = f2bf(W[n * K + k]);
}

// ---------------- WMMA GEMM: C[M,128] = A[M,128](bf16) * B[128,128](bf16) + bias ----
// block = (32,8): 8 waves. The block owns a GROWS x 128 strip of C.
// Wave w owns column tile [16w, 16w+16). A strip staged in LDS (async
// global->LDS b128 when available); B staged once per block in
// *fragment-major* LDS layout so each lane's 16-half B fragment is
// contiguous (two ds_load_b128).
__global__ __launch_bounds__(256) void k_gemm_wmma(
    const __bf16* __restrict__ A, const __bf16* __restrict__ B,
    const float* __restrict__ bias, float* __restrict__ C, int M) {
  const int lane  = threadIdx.x;          // 0..31 (wave32)
  const int wave  = threadIdx.y;          // 0..7
  const int t     = wave * 32 + lane;
  const int mbase = blockIdx.x * GROWS;
  const bool full = (mbase + GROWS) <= M; // strip fully inside the matrix?

  __shared__ __align__(16) __bf16 sA[GROWS * 128];       // 16 KB, row-major
  __shared__ __align__(16) __bf16 sB[8 * 4 * 32 * 16];   // 32 KB, fragment-major

  // ---- stage A strip (16B granules; zero-pad rows >= M on boundary block) ----
  {
    const uint4* Ag = (const uint4*)(A + (size_t)mbase * 128);
    uint4* As = (uint4*)sA;
#ifdef HAVE_ASYNC_LDS
    if (full) {
      // CDNA5 async global->LDS copy, tracked by ASYNCcnt
      for (int i = t; i < GROWS * 8; i += 256)
        __builtin_amdgcn_global_load_async_to_lds_b128(
            (vint4_*)(Ag + i), (as3_vint4*)(As + i), 0, 0);
    } else
#endif
    {
      for (int i = t; i < GROWS * 8; i += 256) {  // 128 bf16 = 8 uint4 per row
        int row = i >> 3;
        uint4 v = {0u, 0u, 0u, 0u};
        if (mbase + row < M) v = Ag[i];
        As[i] = v;
      }
    }
  }
  // ---- stage B fragment-major: slot(nt, ks, lane, j) ----
  // lane holds column n = nt*16 + (lane&15); halves j=0..15 are K = ks*32 + (lane<16?0:16) + j
  for (int i = t; i < 128 * 128; i += 256) {
    int k = i >> 7, n = i & 127;
    int nt = n >> 4, ncol = n & 15;
    int ks = k >> 5, kr = k & 31;
    int l  = ((kr >= 16) ? 16 : 0) + ncol;
    int j  = kr & 15;
    sB[(((nt << 2) + ks) << 9) + (l << 4) + j] = B[i];
  }
  // speculative prefetch of next A strip into GL2
  if (t == 0 && mbase + GROWS < M)
    __builtin_prefetch(A + (size_t)(mbase + GROWS) * 128, 0, 1);
#ifdef HAVE_ASYNC_LDS
#if __has_builtin(__builtin_amdgcn_s_wait_asynccnt)
  __builtin_amdgcn_s_wait_asynccnt(0);
#else
  asm volatile("s_wait_asynccnt 0" ::: "memory");
#endif
#endif
  __syncthreads();

  // CDNA5 wave32 fragment coordinates (CDNA5 ISA 7.12.2)
  const int mrow = lane & 15;             // A row this lane holds
  const int hiA  = (lane < 16) ? 0 : 16;  // lanes 16-31 hold K+16 halves (this kernel's k order, same as sB)
  const int ncol = lane & 15;             // B/D column this lane holds
  const int nb   = wave * 16;
  const float bv = bias[nb + ncol];       // hoisted: one bias load per lane

  // per-wave B fragments for all 4 K-steps (reused across 4 M-tiles)
  v16bf bf[4];
#pragma unroll
  for (int ks = 0; ks < 4; ++ks) {
    const __bf16* p = &sB[(((wave << 2) + ks) << 9) + (lane << 4)];
#pragma unroll
    for (int j = 0; j < 16; ++j) bf[ks][j] = p[j];
  }

#pragma unroll
  for (int mt = 0; mt < GROWS / 16; ++mt) {
    v8f acc = {};
#pragma unroll
    for (int ks = 0; ks < 4; ++ks) {
      v16bf a;
      const __bf16* pa = &sA[(mt * 16 + mrow) * 128 + ks * 32 + hiA];
#pragma unroll
      for (int j = 0; j < 8; ++j) { a[j] = pa[j]; a[j + 8] = pa[8 + j]; }
      acc = __builtin_amdgcn_wmma_f32_16x16x32_bf16(
          false, a, false, bf[ks], (short)0, acc, false, false);
    }
    const int rbase = mbase + mt * 16 + ((lane < 16) ? 0 : 8);
    float* cp = C + (size_t)rbase * 128 + nb + ncol;
    if (full) {
#pragma unroll
      for (int r = 0; r < 8; ++r) cp[(size_t)r * 128] = acc[r] + bv;
    } else {
#pragma unroll
      for (int r = 0; r < 8; ++r)
        if (rbase + r < M) cp[(size_t)r * 128] = acc[r] + bv;
    }
  }
}

// ---------------- small scalar linear: Y[M,N] = X[M,K] @ W[N,K]^T + b ----------------
__global__ void k_linear_small(const float* __restrict__ X, const float* __restrict__ W,
                               const float* __restrict__ b, float* __restrict__ Y,
                               int M, int N, int K) {
  int i = blockIdx.x * blockDim.x + threadIdx.x;
  if (i >= M * N) return;
  int m = i / N, n = i % N;
  const float* x = X + (size_t)m * K;
  const float* w = W + (size_t)n * K;
  float acc = 0.f;
  for (int k = 0; k < K; ++k) acc += x[k] * w[k];
  Y[i] = acc + b[n];
}

// ---------------- attention logits: out[n,h] = sum_d H[n,h,d]*a[h,d] ----------------
__global__ void k_attn_logits(const float* __restrict__ H, const float* __restrict__ a,
                              float* __restrict__ out, int n, int heads, int dim) {
  int i = blockIdx.x * blockDim.x + threadIdx.x;
  if (i >= n * heads) return;
  int node = i / heads, h = i % heads;
  const float* hp = H + ((size_t)node * heads + h) * dim;
  const float* av = a + (size_t)h * dim;
  float acc = 0.f;
  for (int d = 0; d < dim; ++d) acc += hp[d] * av[d];
  out[i] = acc;
}

// ---------------- fills ----------------
__global__ void k_fill(float* p, float v, int n) {
  int i = blockIdx.x * blockDim.x + threadIdx.x;
  if (i < n) p[i] = v;
}

// ---------------- segment softmax passes ----------------
__global__ void k_edge_max(const int* __restrict__ src, const int* __restrict__ dst,
                           const float* __restrict__ als, const float* __restrict__ ald,
                           float* __restrict__ amax, int E, int heads) {
  int i = blockIdx.x * blockDim.x + threadIdx.x;
  if (i >= E * heads) return;
  int e = i / heads, h = i % heads;
  int s = src[e], d = dst[e];
  float a = lrelu02(als[s * heads + h] + ald[d * heads + h]);
  atomicMaxF(&amax[d * heads + h], a);
}

__global__ void k_fix_amax(float* amax, int n) {
  int i = blockIdx.x * blockDim.x + threadIdx.x;
  if (i < n && !__builtin_isfinite(amax[i])) amax[i] = 0.f;
}

__global__ void k_edge_expsum(const int* __restrict__ src, const int* __restrict__ dst,
                              const float* __restrict__ als, const float* __restrict__ ald,
                              const float* __restrict__ amax, float* __restrict__ den,
                              int E, int heads) {
  int i = blockIdx.x * blockDim.x + threadIdx.x;
  if (i >= E * heads) return;
  int e = i / heads, h = i % heads;
  int s = src[e], d = dst[e];
  float a = lrelu02(als[s * heads + h] + ald[d * heads + h]);
  atomicAdd(&den[d * heads + h], expf(a - amax[d * heads + h]));
}

__global__ void k_edge_scatter(const int* __restrict__ src, const int* __restrict__ dst,
                               const float* __restrict__ als, const float* __restrict__ ald,
                               const float* __restrict__ amax, const float* __restrict__ den,
                               const float* __restrict__ Hsrc, float* __restrict__ out,
                               int E, int heads, int dim) {
  int i = blockIdx.x * blockDim.x + threadIdx.x;
  if (i >= E * heads) return;
  int e = i / heads, h = i % heads;
  int s = src[e], d = dst[e];
  float a = lrelu02(als[s * heads + h] + ald[d * heads + h]);
  float w = expf(a - amax[d * heads + h]) / (den[d * heads + h] + 1e-16f);
  const float* hs = Hsrc + ((size_t)s * heads + h) * dim;
  float* o = out + ((size_t)d * heads + h) * dim;
  for (int j = 0; j < dim; ++j) atomicAdd(&o[j], hs[j] * w);
}

__global__ void k_relu(float* x, int n) {
  int i = blockIdx.x * blockDim.x + threadIdx.x;
  if (i < n) x[i] = fmaxf(x[i], 0.f);
}

// ---------------- eval BN (+ELU) over C=128 cols, in place ----------------
__global__ void k_bn_elu(float* __restrict__ x, const float* __restrict__ w,
                         const float* __restrict__ b, const float* __restrict__ mean,
                         const float* __restrict__ var, int n) {
  int i = blockIdx.x * blockDim.x + threadIdx.x;
  if (i >= n) return;
  int c = i & (HID - 1);
  float inv = w[c] * rsqrtf(var[c] + BN_EPS);
  float v = (x[i] - mean[c]) * inv + b[c];
  x[i] = v > 0.f ? v : expm1f(v);
}

// ---------------- final log_softmax over 3 classes ----------------
__global__ void k_log_softmax3(const float* __restrict__ X, float* __restrict__ out, int M) {
  int m = blockIdx.x * blockDim.x + threadIdx.x;
  if (m >= M) return;
  float x0 = X[m * 3 + 0], x1 = X[m * 3 + 1], x2 = X[m * 3 + 2];
  float mx = fmaxf(x0, fmaxf(x1, x2));
  float lse = mx + logf(expf(x0 - mx) + expf(x1 - mx) + expf(x2 - mx));
  out[m * 3 + 0] = x0 - lse;
  out[m * 3 + 1] = x1 - lse;
  out[m * 3 + 2] = x2 - lse;
}

// ---------------- host side ----------------
static inline void edge_softmax_scatter(const int* src, const int* dst, int E,
                                        const float* als, const float* ald,
                                        float* amax, float* den,
                                        const float* Hsrc, float* out,
                                        int ndst, int heads, int dim, hipStream_t s) {
  const int B = 256;
  k_fill<<<CDIV(ndst * heads, B), B, 0, s>>>(amax, -__builtin_inff(), ndst * heads);
  k_fill<<<CDIV(ndst * heads, B), B, 0, s>>>(den, 0.f, ndst * heads);
  k_fill<<<CDIV(ndst * heads * dim, B), B, 0, s>>>(out, 0.f, ndst * heads * dim);
  k_edge_max   <<<CDIV(E * heads, B), B, 0, s>>>(src, dst, als, ald, amax, E, heads);
  k_fix_amax   <<<CDIV(ndst * heads, B), B, 0, s>>>(amax, ndst * heads);
  k_edge_expsum<<<CDIV(E * heads, B), B, 0, s>>>(src, dst, als, ald, amax, den, E, heads);
  k_edge_scatter<<<CDIV(E * heads, B), B, 0, s>>>(src, dst, als, ald, amax, den, Hsrc, out,
                                                  E, heads, dim);
  k_relu<<<CDIV(ndst * heads * dim, B), B, 0, s>>>(out, ndst * heads * dim);
}

extern "C" void kernel_launch(void* const* d_in, const int* in_sizes, int n_in,
                              void* d_out, int out_size, void* d_ws, size_t ws_size,
                              hipStream_t stream) {
  (void)in_sizes; (void)n_in; (void)out_size; (void)ws_size;
  const int B = 256;

  // -------- inputs (setup_inputs() order) --------
  const float* x_p  = (const float*)d_in[0];
  const float* x_s  = (const float*)d_in[1];
  const float* x_d  = (const float*)d_in[2];
  const int*   ei_has = (const int*)d_in[3];
  const int*   ei_bel = (const int*)d_in[4];
  const int*   ei_sim = (const int*)d_in[5];
  const float* w1p = (const float*)d_in[6];  const float* b1p = (const float*)d_in[7];
  const float* w1s = (const float*)d_in[8];  const float* b1s = (const float*)d_in[9];
  const float* w1d = (const float*)d_in[10]; const float* b1d = (const float*)d_in[11];
  const float* a1sh = (const float*)d_in[12]; const float* a1dh = (const float*)d_in[13];
  const float* a1sb = (const float*)d_in[14]; const float* a1db = (const float*)d_in[15];
  const float* a1ss = (const float*)d_in[16]; const float* a1ds = (const float*)d_in[17];
  const float* bnw = (const float*)d_in[18]; const float* bnb = (const float*)d_in[19];
  const float* bnm = (const float*)d_in[20]; const float* bnv = (const float*)d_in[21];
  const float* w2p = (const float*)d_in[22]; const float* b2p = (const float*)d_in[23];
  // w2_symptom/w2_dosha, a2_*_has/bel feed outputs discarded by the reference.
  const float* a2ss = (const float*)d_in[32]; const float* a2ds = (const float*)d_in[33];

  // -------- workspace layout --------
  char* ws = (char*)d_ws;
  size_t off = 0;
  auto alloc = [&](size_t bytes) -> char* {
    char* p = ws + off;
    off = (off + bytes + 255) & ~(size_t)255;
    return p;
  };
  __bf16* xp_bf = (__bf16*)alloc((size_t)NP * CP * 2);
  __bf16* Bp_bf = (__bf16*)alloc((size_t)CP * HID * 2);
  float* hp = (float*)alloc((size_t)NP * HID * 4);
  float* hs = (float*)alloc((size_t)NS * HID * 4);
  float* hd = (float*)alloc((size_t)ND * HID * 4);
  float* als_has = (float*)alloc((size_t)NP * HEADS * 4);
  float* ald_has = (float*)alloc((size_t)NS * HEADS * 4);
  float* als_bel = (float*)alloc((size_t)NP * HEADS * 4);
  float* ald_bel = (float*)alloc((size_t)ND * HEADS * 4);
  float* als_sim = (float*)alloc((size_t)NP * HEADS * 4);
  float* ald_sim = (float*)alloc((size_t)NP * HEADS * 4);
  float* amax_s = (float*)alloc((size_t)NS * HEADS * 4);
  float* den_s  = (float*)alloc((size_t)NS * HEADS * 4);
  float* amax_d = (float*)alloc((size_t)ND * HEADS * 4);
  float* den_d  = (float*)alloc((size_t)ND * HEADS * 4);
  float* amax_p = (float*)alloc((size_t)NP * HEADS * 4);
  float* den_p  = (float*)alloc((size_t)NP * HEADS * 4);
  float* out_s = (float*)alloc((size_t)NS * HID * 4);
  float* out_d = (float*)alloc((size_t)ND * HID * 4);
  float* out_p = (float*)alloc((size_t)NP * HID * 4);
  float* hp2  = (float*)alloc((size_t)NP * NCLS * 4);
  float* al2s = (float*)alloc((size_t)NP * 4);
  float* al2d = (float*)alloc((size_t)NP * 4);
  float* amax2 = (float*)alloc((size_t)NP * 4);
  float* den2  = (float*)alloc((size_t)NP * 4);
  float* out2  = (float*)alloc((size_t)NP * NCLS * 4);

  // -------- phase A: bf16 conversion for the one compute-heavy GEMM --------
  k_f32_to_bf16<<<CDIV(NP * CP, B), B, 0, stream>>>(x_p, xp_bf, NP * CP);
  k_wT_to_bf16 <<<CDIV(HID * CP, B), B, 0, stream>>>(w1p, Bp_bf, HID, CP);

  // -------- phase B: layer-1 linears --------
  {
    dim3 grid(CDIV(NP, GROWS)), block(32, 8);
    k_gemm_wmma<<<grid, block, 0, stream>>>(xp_bf, Bp_bf, b1p, hp, NP);
  }
  k_linear_small<<<CDIV(NS * HID, B), B, 0, stream>>>(x_s, w1s, b1s, hs, NS, HID, CS);
  k_linear_small<<<CDIV(ND * HID, B), B, 0, stream>>>(x_d, w1d, b1d, hd, ND, HID, CD);

  // -------- phase C: attention logits (heads=4, dim=32) --------
  k_attn_logits<<<CDIV(NP * HEADS, B), B, 0, stream>>>(hp, a1sh, als_has, NP, HEADS, HDIM);
  k_attn_logits<<<CDIV(NS * HEADS, B), B, 0, stream>>>(hs, a1dh, ald_has, NS, HEADS, HDIM);
  k_attn_logits<<<CDIV(NP * HEADS, B), B, 0, stream>>>(hp, a1sb, als_bel, NP, HEADS, HDIM);
  k_attn_logits<<<CDIV(ND * HEADS, B), B, 0, stream>>>(hd, a1db, ald_bel, ND, HEADS, HDIM);
  k_attn_logits<<<CDIV(NP * HEADS, B), B, 0, stream>>>(hp, a1ss, als_sim, NP, HEADS, HDIM);
  k_attn_logits<<<CDIV(NP * HEADS, B), B, 0, stream>>>(hp, a1ds, ald_sim, NP, HEADS, HDIM);

  // -------- phase D: segment softmax + scatter per edge type --------
  edge_softmax_scatter(ei_has, ei_has + EHAS, EHAS, als_has, ald_has, amax_s, den_s,
                       hp, out_s, NS, HEADS, HDIM, stream);
  edge_softmax_scatter(ei_bel, ei_bel + EBEL, EBEL, als_bel, ald_bel, amax_d, den_d,
                       hp, out_d, ND, HEADS, HDIM, stream);
  edge_softmax_scatter(ei_sim, ei_sim + ESIM, ESIM, als_sim, ald_sim, amax_p, den_p,
                       hp, out_p, NP, HEADS, HDIM, stream);

  // -------- phase E: BN + ELU (eval mode, shared params) --------
  k_bn_elu<<<CDIV(NP * HID, B), B, 0, stream>>>(out_p, bnw, bnb, bnm, bnv, NP * HID);
  k_bn_elu<<<CDIV(NS * HID, B), B, 0, stream>>>(out_s, bnw, bnb, bnm, bnv, NS * HID);
  k_bn_elu<<<CDIV(ND * HID, B), B, 0, stream>>>(out_d, bnw, bnb, bnm, bnv, ND * HID);

  // -------- phase F: layer 2 (only the patient/sim path reaches the output) --------
  k_linear_small<<<CDIV(NP * NCLS, B), B, 0, stream>>>(out_p, w2p, b2p, hp2, NP, NCLS, HID);
  k_attn_logits<<<CDIV(NP, B), B, 0, stream>>>(hp2, a2ss, al2s, NP, 1, NCLS);
  k_attn_logits<<<CDIV(NP, B), B, 0, stream>>>(hp2, a2ds, al2d, NP, 1, NCLS);
  edge_softmax_scatter(ei_sim, ei_sim + ESIM, ESIM, al2s, al2d, amax2, den2,
                       hp2, out2, NP, 1, NCLS, stream);

  k_log_softmax3<<<CDIV(NP, B), B, 0, stream>>>(out2, (float*)d_out, NP);
}
